// LinearPredictionHead_29789893165417
// MI455X (gfx1250) — matrix operand order repaired and checked by reading, not compiled, not run on hardware
//
#include <hip/hip_runtime.h>
#include <hip/hip_bf16.h>

typedef __attribute__((ext_vector_type(16))) __bf16 v16bf;
typedef __attribute__((ext_vector_type(8)))  __bf16 v8bf;
typedef __attribute__((ext_vector_type(8)))  float  v8f;

#define NE 4
#define DM 1024
#define PL 720
#define OUT_C 32
#define BC 2048

// ---------------------------------------------------------------------------
// Pack pass 1: A[e][row][d] = bf16(xs_e[row, L_e-1, d])   (16 MB)
// ---------------------------------------------------------------------------
__global__ __launch_bounds__(256)
void pack_A_kernel(const float* __restrict__ xs0, const float* __restrict__ xs1,
                   const float* __restrict__ xs2, const float* __restrict__ xs3,
                   __bf16* __restrict__ dstA)
{
    const int e = blockIdx.y;
    const float* src;
    int L;
    switch (e) {
        case 0:  src = xs0; L = 32; break;
        case 1:  src = xs1; L = 16; break;
        case 2:  src = xs2; L = 8;  break;
        default: src = xs3; L = 4;  break;
    }
    size_t t = (size_t)blockIdx.x * 256 + threadIdx.x;  // 0 .. 2048*128-1
    int r  = (int)(t >> 7);        // row 0..2047 (128 threads/row)
    int d0 = ((int)t & 127) * 8;   // 0..1016
    const float* sp = src + ((size_t)r * L + (L - 1)) * DM + d0;
    float4 x0 = ((const float4*)sp)[0];
    float4 x1 = ((const float4*)sp)[1];
    v8bf o;
    o[0] = (__bf16)x0.x; o[1] = (__bf16)x0.y; o[2] = (__bf16)x0.z; o[3] = (__bf16)x0.w;
    o[4] = (__bf16)x1.x; o[5] = (__bf16)x1.y; o[6] = (__bf16)x1.z; o[7] = (__bf16)x1.w;
    *(v8bf*)(dstA + ((size_t)e * BC + r) * DM + d0) = o;
}

// ---------------------------------------------------------------------------
// Pack pass 2: WT[e][p][d] = bf16(W_e[d][p])   (transpose, 5.8 MB)
// Reads are coalesced across lanes (p fastest); one-time scattered stores.
// ---------------------------------------------------------------------------
__global__ __launch_bounds__(256)
void pack_WT_kernel(const float* __restrict__ W0, const float* __restrict__ W1,
                    const float* __restrict__ W2, const float* __restrict__ W3,
                    __bf16* __restrict__ dstWT)
{
    const int e = blockIdx.y;
    const float* W;
    switch (e) {
        case 0:  W = W0; break;
        case 1:  W = W1; break;
        case 2:  W = W2; break;
        default: W = W3; break;
    }
    size_t t = (size_t)blockIdx.x * 256 + threadIdx.x;  // 0 .. 128*128-1
    int d0 = (int)(t >> 7) * 8;   // d block 0..1016 (slow)
    int p  = ((int)t & 127);      // partial p; need full 720 coverage:
    p += (int)blockIdx.z * 128;   // grid.z covers p in chunks of 128
    if (p >= PL) return;
    float v[8];
#pragma unroll
    for (int j = 0; j < 8; ++j)
        v[j] = W[(size_t)(d0 + j) * PL + p];
    v8bf o;
#pragma unroll
    for (int j = 0; j < 8; ++j)
        o[j] = (__bf16)v[j];
    *(v8bf*)(dstWT + ((size_t)e * PL + p) * DM + d0) = o;
}

// ---------------------------------------------------------------------------
// Main GEMM on packed bf16: one wave = one 16x16 (bc x p) tile, 4 experts.
// No LDS, no barriers, no converts: pure b128 loads + v_wmma.
// ---------------------------------------------------------------------------
__global__ __launch_bounds__(256)
void moe_gemm_packed_kernel(const __bf16* __restrict__ A, const __bf16* __restrict__ WT,
                            const float* __restrict__ gates,
                            const float* __restrict__ bi0, const float* __restrict__ bi1,
                            const float* __restrict__ bi2, const float* __restrict__ bi3,
                            float* __restrict__ out)
{
    const int tid  = threadIdx.x;
    const int wid  = tid >> 5;
    const int lane = tid & 31;
    const int hlf  = lane >> 4;
    const int l16  = lane & 15;

    const int tile_n = blockIdx.x >> 4;                 // 0..44
    const int tile_m = ((blockIdx.x & 15) << 3) | wid;  // 0..127
    const int n_base = tile_n * 16;
    const int m_base = tile_m * 16;

    const float* bs[NE] = {bi0, bi1, bi2, bi3};

    // Per-expert row pointers (A) and column-row pointers (WT).
    const __bf16* arow[NE];
    const __bf16* brow[NE];
#pragma unroll
    for (int e = 0; e < NE; ++e) {
        arow[e] = A  + ((size_t)e * BC + m_base + l16) * DM;
        brow[e] = WT + ((size_t)e * PL + n_base + l16) * DM;
    }

    v8f zero = {0.f, 0.f, 0.f, 0.f, 0.f, 0.f, 0.f, 0.f};
    v8f acc[NE] = {zero, zero, zero, zero};

    for (int k0 = 0; k0 < DM; k0 += 32) {
#pragma unroll
        for (int e = 0; e < NE; ++e) {
            // A fragment (bf16 16x32): lanes 0-15 K 0..15 of row M=l16,
            // lanes 16-31 K 16..31; two contiguous 16B loads.
            v8bf a0 = *(const v8bf*)(arow[e] + k0 + hlf * 16);
            v8bf a1 = *(const v8bf*)(arow[e] + k0 + hlf * 16 + 8);
            v16bf a = __builtin_shufflevector(a0, a1,
                0, 1, 2, 3, 4, 5, 6, 7, 8, 9, 10, 11, 12, 13, 14, 15);

            // B fragment (bf16 32x16): column N=l16; lanes 0-15 K=0..15,
            // lanes 16-31 K=16..31; one contiguous 32B load from WT.
            v16bf bfr = *(const v16bf*)(brow[e] + k0 + hlf * 16);

            acc[e] = __builtin_amdgcn_wmma_f32_16x16x32_bf16(
                false, a, false, bfr, (short)0, acc[e], false, false);
        }
        if (k0 + 32 < DM) {
#pragma unroll
            for (int e = 0; e < NE; ++e)
                __builtin_prefetch(arow[e] + k0 + 32 + hlf * 8, 0, 1);
        }
    }

    // Epilogue: gated log-sum-exp combine + [B,P,C] transpose.
    const int b_idx  = m_base >> 5;
    const int c_base = (m_base & 31) + hlf * 8;
    const int p      = n_base + l16;

    float g[NE], bia[NE];
#pragma unroll
    for (int e = 0; e < NE; ++e) {
        g[e]   = gates[b_idx * NE + e];
        bia[e] = bs[e][p];
    }

    float vals[8];
#pragma unroll
    for (int i = 0; i < 8; ++i) {
        float s = 0.f;
#pragma unroll
        for (int e = 0; e < NE; ++e)
            s += g[e] * __expf(acc[e][i] + bia[e]);
        if (s == 0.f) s = 2.2204460492503131e-16f;  // np double eps
        vals[i] = __logf(s);
    }

    float* op = out + (size_t)b_idx * (PL * OUT_C) + (size_t)p * OUT_C + c_base;
    *(float4*)(op)     = make_float4(vals[0], vals[1], vals[2], vals[3]);
    *(float4*)(op + 4) = make_float4(vals[4], vals[5], vals[6], vals[7]);
}

// ---------------------------------------------------------------------------
// Fallback fused kernel (round-2 version): used if workspace is too small.
// ---------------------------------------------------------------------------
__global__ __launch_bounds__(256)
void moe_head_fused_kernel(const float* __restrict__ xs0, const float* __restrict__ xs1,
                           const float* __restrict__ xs2, const float* __restrict__ xs3,
                           const float* __restrict__ gates,
                           const float* __restrict__ W0, const float* __restrict__ bi0,
                           const float* __restrict__ W1, const float* __restrict__ bi1,
                           const float* __restrict__ W2, const float* __restrict__ bi2,
                           const float* __restrict__ W3, const float* __restrict__ bi3,
                           float* __restrict__ out)
{
    __shared__ __align__(32) __bf16 Bs[2][NE][16][32];

    const int tid = threadIdx.x;
    const int wid = tid >> 5;
    const int lane = tid & 31;
    const int hlf = lane >> 4;
    const int l16 = lane & 15;

    const int tile_n = blockIdx.x >> 4;
    const int tile_m = ((blockIdx.x & 15) << 3) | wid;
    const int n_base = tile_n * 16;
    const int m_base = tile_m * 16;

    const float* Ws[NE] = {W0, W1, W2, W3};
    const float* bs[NE] = {bi0, bi1, bi2, bi3};

    const float* arow[NE];
    {
        const float* xsp[NE] = {xs0, xs1, xs2, xs3};
        const int Ls[NE] = {32, 16, 8, 4};
#pragma unroll
        for (int e = 0; e < NE; ++e) {
            const int L = Ls[e];
            arow[e] = xsp[e] + ((size_t)(m_base + l16) * L + (L - 1)) * DM;
        }
    }

    const int fn = tid & 15;
    const int fk = tid >> 4;

    v8f zero = {0.f, 0.f, 0.f, 0.f, 0.f, 0.f, 0.f, 0.f};
    v8f acc[NE] = {zero, zero, zero, zero};

    for (int k0 = 0; k0 < DM; k0 += 32) {
        const int buf = (k0 >> 5) & 1;

        float4 ar[NE][4];
#pragma unroll
        for (int e = 0; e < NE; ++e) {
            const float4* pa = (const float4*)(arow[e] + k0 + hlf * 16);
            ar[e][0] = pa[0];
            ar[e][1] = pa[1];
            ar[e][2] = pa[2];
            ar[e][3] = pa[3];
        }
        {
            float w0[NE], w1[NE];
#pragma unroll
            for (int e = 0; e < NE; ++e) {
                w0[e] = Ws[e][(size_t)(k0 + fk) * PL + (n_base + fn)];
                w1[e] = Ws[e][(size_t)(k0 + fk + 16) * PL + (n_base + fn)];
            }
#pragma unroll
            for (int e = 0; e < NE; ++e) {
                Bs[buf][e][fn][fk]      = (__bf16)w0[e];
                Bs[buf][e][fn][fk + 16] = (__bf16)w1[e];
            }
        }
        __syncthreads();

#pragma unroll
        for (int e = 0; e < NE; ++e) {
            v16bf bf = *(const v16bf*)&Bs[buf][e][l16][hlf * 16];
            v16bf a;
            a[0]  = (__bf16)ar[e][0].x; a[1]  = (__bf16)ar[e][0].y;
            a[2]  = (__bf16)ar[e][0].z; a[3]  = (__bf16)ar[e][0].w;
            a[4]  = (__bf16)ar[e][1].x; a[5]  = (__bf16)ar[e][1].y;
            a[6]  = (__bf16)ar[e][1].z; a[7]  = (__bf16)ar[e][1].w;
            a[8]  = (__bf16)ar[e][2].x; a[9]  = (__bf16)ar[e][2].y;
            a[10] = (__bf16)ar[e][2].z; a[11] = (__bf16)ar[e][2].w;
            a[12] = (__bf16)ar[e][3].x; a[13] = (__bf16)ar[e][3].y;
            a[14] = (__bf16)ar[e][3].z; a[15] = (__bf16)ar[e][3].w;

            acc[e] = __builtin_amdgcn_wmma_f32_16x16x32_bf16(
                false, a, false, bf, (short)0, acc[e], false, false);
        }
    }

    const int b_idx  = m_base >> 5;
    const int c_base = (m_base & 31) + hlf * 8;
    const int p      = n_base + l16;

    float g[NE], bia[NE];
#pragma unroll
    for (int e = 0; e < NE; ++e) {
        g[e]   = gates[b_idx * NE + e];
        bia[e] = bs[e][p];
    }

    float vals[8];
#pragma unroll
    for (int i = 0; i < 8; ++i) {
        float s = 0.f;
#pragma unroll
        for (int e = 0; e < NE; ++e)
            s += g[e] * __expf(acc[e][i] + bia[e]);
        if (s == 0.f) s = 2.2204460492503131e-16f;
        vals[i] = __logf(s);
    }

    float* op = out + (size_t)b_idx * (PL * OUT_C) + (size_t)p * OUT_C + c_base;
    *(float4*)(op)     = make_float4(vals[0], vals[1], vals[2], vals[3]);
    *(float4*)(op + 4) = make_float4(vals[4], vals[5], vals[6], vals[7]);
}

// ---------------------------------------------------------------------------
extern "C" void kernel_launch(void* const* d_in, const int* in_sizes, int n_in,
                              void* d_out, int out_size, void* d_ws, size_t ws_size,
                              hipStream_t stream) {
    (void)in_sizes; (void)n_in; (void)out_size;
    const float* xs0   = (const float*)d_in[0];
    const float* xs1   = (const float*)d_in[1];
    const float* xs2   = (const float*)d_in[2];
    const float* xs3   = (const float*)d_in[3];
    const float* gates = (const float*)d_in[4];
    const float* W0    = (const float*)d_in[5];
    const float* b0    = (const float*)d_in[6];
    const float* W1    = (const float*)d_in[7];
    const float* b1    = (const float*)d_in[8];
    const float* W2    = (const float*)d_in[9];
    const float* b2    = (const float*)d_in[10];
    const float* W3    = (const float*)d_in[11];
    const float* b3    = (const float*)d_in[12];
    float* out = (float*)d_out;

    const size_t elemsA = (size_t)NE * BC * DM;  // 8.4M bf16
    const size_t elemsW = (size_t)NE * PL * DM;  // 2.9M bf16
    const size_t need   = (elemsA + elemsW) * 2; // bytes

    if (ws_size >= need && d_ws != nullptr) {
        __bf16* wsA  = (__bf16*)d_ws;
        __bf16* wsWT = wsA + elemsA;

        // Pack A: 2048*128 threads per expert
        pack_A_kernel<<<dim3(BC * 128 / 256, NE), dim3(256), 0, stream>>>(
            xs0, xs1, xs2, xs3, wsA);
        // Pack WT: (d-blocks=128) x (p-chunks: 6 of 128 covers 720), per expert
        pack_WT_kernel<<<dim3(128 * 128 / 256, NE, 6), dim3(256), 0, stream>>>(
            W0, W1, W2, W3, wsWT);
        // GEMM + combine
        moe_gemm_packed_kernel<<<dim3(45 * 16), dim3(256), 0, stream>>>(
            wsA, wsWT, gates, b0, b1, b2, b3, out);
    } else {
        moe_head_fused_kernel<<<dim3(45 * 16), dim3(256), 0, stream>>>(
            xs0, xs1, xs2, xs3, gates, W0, b0, W1, b1, W2, b2, W3, b3, out);
    }
}
